// CrossAttention_11914239279792
// MI455X (gfx1250) — matrix operand order repaired and checked by reading, not compiled, not run on hardware
//
#include <hip/hip_runtime.h>
#include <stdint.h>

// ---------------------------------------------------------------- types
typedef __bf16 bf16;
typedef __attribute__((ext_vector_type(16))) __bf16 v16bf;
typedef __attribute__((ext_vector_type(8)))  __bf16 bf16x8;
typedef __attribute__((ext_vector_type(8)))  float  v8f;
typedef __attribute__((ext_vector_type(4)))  float  f32x4;

#define BATCH  2
#define NQ     2048
#define NKV    2048
#define QDIM   1024
#define HEADS  16
#define DHEAD  64
#define INNER  1024
#define SCALE  0.125f            // 64^-0.5
#define NEGMAX (-3.402823466e38f)

union ABFrag { v16bf v; bf16x8 h[2]; };

static __device__ inline v8f wmma_bf16(const ABFrag& a, const ABFrag& b, v8f c) {
  // D = A(16x32 bf16) * B(32x16 bf16) + C(16x16 f32)
  return __builtin_amdgcn_wmma_f32_16x16x32_bf16(false, a.v, false, b.v,
                                                 (short)0, c, false, false);
}

// ---------------------------------------------------------------- GEMM
// C[M,N] = A[M,K] * W[N,K]^T   (torch Linear semantics)
// MODE 0: write bf16, permuted to [B, H, seq, DHEAD]  (Q/K/V projections)
// MODE 1: write f32 + bias, plain row-major            (output projection)
#define BM 128
#define BN 128
#define BK 32
#define LDT (BK + 8)   // LDS row pad: 80B rows -> conflict-free frag reads

static __device__ inline void tile_row16(const float* __restrict__ src, bf16* dst) {
  f32x4 t0 = *(const f32x4*)(src + 0);
  f32x4 t1 = *(const f32x4*)(src + 4);
  f32x4 t2 = *(const f32x4*)(src + 8);
  f32x4 t3 = *(const f32x4*)(src + 12);
  bf16x8 o0, o1;
  o0[0]=(bf16)t0[0]; o0[1]=(bf16)t0[1]; o0[2]=(bf16)t0[2]; o0[3]=(bf16)t0[3];
  o0[4]=(bf16)t1[0]; o0[5]=(bf16)t1[1]; o0[6]=(bf16)t1[2]; o0[7]=(bf16)t1[3];
  o1[0]=(bf16)t2[0]; o1[1]=(bf16)t2[1]; o1[2]=(bf16)t2[2]; o1[3]=(bf16)t2[3];
  o1[4]=(bf16)t3[0]; o1[5]=(bf16)t3[1]; o1[6]=(bf16)t3[2]; o1[7]=(bf16)t3[3];
  *(bf16x8*)dst       = o0;
  *(bf16x8*)(dst + 8) = o1;
}
static __device__ inline void tile_row16(const bf16* __restrict__ src, bf16* dst) {
  *(bf16x8*)dst       = *(const bf16x8*)(src);
  *(bf16x8*)(dst + 8) = *(const bf16x8*)(src + 8);
}

template<typename AT, int MODE>
__global__ __launch_bounds__(256)
void gemm_bf16_wmma(const AT* __restrict__ A, const float* __restrict__ W,
                    const float* __restrict__ bias,
                    bf16* __restrict__ outb, float* __restrict__ outf,
                    int M, int K, int N)
{
  __shared__ __align__(16) bf16 As[BM][LDT];
  __shared__ __align__(16) bf16 Bs[BN][LDT];

  const int tid  = threadIdx.x;
  const int wave = tid >> 5;
  const int lane = tid & 31;
  const int l16  = lane & 15;
  const int hi   = lane >> 4;            // lane half (wave32)

  const int m0 = blockIdx.y * BM;
  const int n0 = blockIdx.x * BN;
  const int wm = (wave & 3) * 32;        // 4 waves along M
  const int wn = (wave >> 2) * 64;       // 2 waves along N

  const int lrow = tid >> 1;             // 0..127: tile row to stage
  const int lcol = (tid & 1) * 16;       // 16 elems per thread

  v8f acc[2][4];
  #pragma unroll
  for (int i = 0; i < 2; i++)
    #pragma unroll
    for (int j = 0; j < 4; j++)
      acc[i][j] = (v8f){0.f,0.f,0.f,0.f,0.f,0.f,0.f,0.f};

  for (int k0 = 0; k0 < K; k0 += BK) {
    tile_row16(A + (size_t)(m0 + lrow) * K + k0 + lcol, &As[lrow][lcol]);
    tile_row16(W + (size_t)(n0 + lrow) * K + k0 + lcol, &Bs[lrow][lcol]);
    if (k0 + BK < K) {   // global_prefetch_b8 of next K-step tiles
      __builtin_prefetch(A + (size_t)(m0 + lrow) * K + k0 + BK + lcol, 0, 1);
      __builtin_prefetch(W + (size_t)(n0 + lrow) * K + k0 + BK + lcol, 0, 1);
    }
    __syncthreads();

    // A frag (16x32): lane holds row M=l16; K halves {hi*8..+7} and {16+hi*8..}
    ABFrag afr[2], bfr[4];
    #pragma unroll
    for (int i = 0; i < 2; i++) {
      const bf16* p = &As[wm + i*16 + l16][hi * 8];
      afr[i].h[0] = *(const bf16x8*)(p);
      afr[i].h[1] = *(const bf16x8*)(p + 16);
    }
    // B frag (32x16): lane holds col N=l16; K halves {hi*8..+7} and {16+hi*8..}
    #pragma unroll
    for (int j = 0; j < 4; j++) {
      const bf16* p = &Bs[wn + j*16 + l16][hi * 8];
      bfr[j].h[0] = *(const bf16x8*)(p);
      bfr[j].h[1] = *(const bf16x8*)(p + 16);
    }
    #pragma unroll
    for (int i = 0; i < 2; i++)
      #pragma unroll
      for (int j = 0; j < 4; j++)
        acc[i][j] = wmma_bf16(afr[i], bfr[j], acc[i][j]);
    __syncthreads();
  }

  // C/D layout: lane = col n (l16), VGPR v = row (hi*8 + v)
  #pragma unroll
  for (int i = 0; i < 2; i++) {
    #pragma unroll
    for (int j = 0; j < 4; j++) {
      const int rbase = m0 + wm + i*16 + hi*8;
      const int c     = n0 + wn + j*16 + l16;
      #pragma unroll
      for (int v = 0; v < 8; v++) {
        const int r = rbase + v;
        const float val = acc[i][j][v];
        if (MODE == 0) {
          const int bb = r >> 11;           // / 2048
          const int ii = r & (NQ - 1);
          const int hh = c >> 6;
          const int dd = c & (DHEAD - 1);
          outb[(((size_t)bb * HEADS + hh) * NQ + ii) * DHEAD + dd] = (bf16)val;
        } else {
          outf[(size_t)r * N + c] = val + bias[c];
        }
      }
    }
  }
}

// ---------------------------------------------------------------- attention
// One block = (b, h, 128 q rows); 8 waves, each owns 16 q rows.
// Flash-style online softmax over kv chunks of 32.
#define QTILE 128
#define KVC   32

__global__ __launch_bounds__(256)
void attn_wmma(const bf16* __restrict__ Q, const bf16* __restrict__ Kmat,
               const bf16* __restrict__ Vmat, const int* __restrict__ tmask,
               const unsigned char* __restrict__ smask, bf16* __restrict__ O)
{
  __shared__ __align__(16) bf16 Ks [KVC][DHEAD + 8];   // [32][72]  kv-major
  __shared__ __align__(16) bf16 Vts[DHEAD][KVC + 8];   // [64][40]  d-major (transposed)
  __shared__ __align__(16) bf16 Ps [8][16][KVC + 8];   // per-wave P relayout pad

  const int tid  = threadIdx.x;
  const int wave = tid >> 5;
  const int lane = tid & 31;
  const int l16  = lane & 15;
  const int hi   = lane >> 4;
  const int mrow = hi * 8;               // this lane's row base in C-layout

  const int qtiles = NQ / QTILE;         // 16
  const int bid = blockIdx.x;
  const int qt  = bid % qtiles;
  const int h   = (bid / qtiles) % HEADS;
  const int b   = bid / (qtiles * HEADS);

  const bf16* Qbase = Q    + (((size_t)b * HEADS + h) * NQ  + qt * QTILE) * DHEAD;
  const bf16* Kbase = Kmat + (((size_t)b * HEADS + h) * NKV) * DHEAD;
  const bf16* Vbase = Vmat + (((size_t)b * HEADS + h) * NKV) * DHEAD;

  const int qr = wave * 16;              // wave's q-row base within tile
  const int qglob0 = qt * QTILE + qr;

  // Q fragments (16x64 split into two 16x32 A frags), loaded once
  ABFrag qf[2];
  {
    const bf16* p = Qbase + (size_t)(qr + l16) * DHEAD;
    #pragma unroll
    for (int i = 0; i < 2; i++) {
      qf[i].h[0] = *(const bf16x8*)(p + i*32 + hi*8);
      qf[i].h[1] = *(const bf16x8*)(p + i*32 + hi*8 + 16);
    }
  }

  v8f oacc[4];
  #pragma unroll
  for (int t = 0; t < 4; t++) oacc[t] = (v8f){0.f,0.f,0.f,0.f,0.f,0.f,0.f,0.f};
  float rmax[8], rsum[8];
  #pragma unroll
  for (int v = 0; v < 8; v++) { rmax[v] = NEGMAX; rsum[v] = 0.f; }

  for (int kv0 = 0; kv0 < NKV; kv0 += KVC) {
    // cooperative stage of K (kv-major) and V (transposed to d-major)
    {
      const int r = tid >> 3;            // 0..31 kv row
      const int c = (tid & 7) * 8;       // 0..56 d col
      *(bf16x8*)&Ks[r][c] = *(const bf16x8*)(Kbase + (size_t)(kv0 + r) * DHEAD + c);
      bf16x8 vv = *(const bf16x8*)(Vbase + (size_t)(kv0 + r) * DHEAD + c);
      #pragma unroll
      for (int e = 0; e < 8; e++) Vts[c + e][r] = vv[e];
    }
    __syncthreads();

    // S = Q * K^T : two 16x16 tiles, reduction over DHEAD=64 (2 WMMAs each)
    v8f s[2];
    #pragma unroll
    for (int nt = 0; nt < 2; nt++) {
      const bf16* p = &Ks[nt*16 + l16][0];
      v8f a = (v8f){0.f,0.f,0.f,0.f,0.f,0.f,0.f,0.f};
      #pragma unroll
      for (int kc = 0; kc < 2; kc++) {
        ABFrag bf_;
        bf_.h[0] = *(const bf16x8*)(p + kc*32 + hi*8);
        bf_.h[1] = *(const bf16x8*)(p + kc*32 + hi*8 + 16);
        a = wmma_bf16(qf[kc], bf_, a);
      }
      s[nt] = a;
    }

    // masks + scale (lane = col j, VGPR v = row)
    const unsigned char* sm = smask + (size_t)b * NKV + kv0;
    const float srcok0 = sm[l16]      ? 1.f : 0.f;
    const float srcok1 = sm[16 + l16] ? 1.f : 0.f;
    const int* tg = tmask + ((size_t)b * NQ + qglob0) * NKV + kv0;
    #pragma unroll
    for (int nt = 0; nt < 2; nt++) {
      const float srcok = nt ? srcok1 : srcok0;
      #pragma unroll
      for (int v = 0; v < 8; v++) {
        const int m = mrow + v;
        const int t = tg[(size_t)m * NKV + nt*16 + l16];
        const bool ok = (t != 0) && (srcok != 0.f);
        s[nt][v] = ok ? s[nt][v] * SCALE : NEGMAX;
      }
    }

    // online softmax: row reductions across the 16 lanes of each half
    float pl[2][8];
    #pragma unroll
    for (int v = 0; v < 8; v++) {
      float cm = fmaxf(s[0][v], s[1][v]);
      #pragma unroll
      for (int off = 8; off; off >>= 1) cm = fmaxf(cm, __shfl_xor(cm, off, 16));
      const float nm   = fmaxf(rmax[v], cm);
      const float corr = __expf(rmax[v] - nm);
      rmax[v] = nm;
      const float p0 = __expf(s[0][v] - nm);
      const float p1 = __expf(s[1][v] - nm);
      pl[0][v] = p0; pl[1][v] = p1;
      float ps = p0 + p1;
      #pragma unroll
      for (int off = 8; off; off >>= 1) ps += __shfl_xor(ps, off, 16);
      rsum[v] = rsum[v] * corr + ps;
      #pragma unroll
      for (int t = 0; t < 4; t++) oacc[t][v] *= corr;
    }

    // relayout P (C-layout -> A-fragment) through per-wave LDS pad
    #pragma unroll
    for (int nt = 0; nt < 2; nt++)
      #pragma unroll
      for (int v = 0; v < 8; v++)
        Ps[wave][mrow + v][nt*16 + l16] = (bf16)pl[nt][v];

    ABFrag pf;
    {
      const bf16* p = &Ps[wave][l16][hi * 8];
      pf.h[0] = *(const bf16x8*)(p);
      pf.h[1] = *(const bf16x8*)(p + 16);
    }
    // O += P(16x32) * V(32x64): 4 WMMAs, B frags from transposed V
    #pragma unroll
    for (int t = 0; t < 4; t++) {
      ABFrag vf;
      const bf16* p = &Vts[t*16 + l16][hi * 8];
      vf.h[0] = *(const bf16x8*)(p);
      vf.h[1] = *(const bf16x8*)(p + 16);
      oacc[t] = wmma_bf16(pf, vf, oacc[t]);
    }
    __syncthreads();
  }

  // normalize and write O as [B, NQ, INNER] bf16
  bf16* Obase = O + ((size_t)b * NQ + qglob0) * INNER + h * DHEAD;
  #pragma unroll
  for (int v = 0; v < 8; v++) {
    const float inv = 1.0f / rsum[v];
    const int m = mrow + v;
    #pragma unroll
    for (int t = 0; t < 4; t++)
      Obase[(size_t)m * INNER + t*16 + l16] = (bf16)(oacc[t][v] * inv);
  }
}

// ---------------------------------------------------------------- launch
extern "C" void kernel_launch(void* const* d_in, const int* in_sizes, int n_in,
                              void* d_out, int out_size, void* d_ws, size_t ws_size,
                              hipStream_t stream)
{
  const float*         x    = (const float*)d_in[0];
  const float*         ctx  = (const float*)d_in[1];
  const int*           tgt  = (const int*)d_in[2];
  const unsigned char* srcm = (const unsigned char*)d_in[3];
  const float*         Wq   = (const float*)d_in[4];
  const float*         Wk   = (const float*)d_in[5];
  const float*         Wv   = (const float*)d_in[6];
  const float*         Wo   = (const float*)d_in[7];
  const float*         bo   = (const float*)d_in[8];
  float*               out  = (float*)d_out;

  const size_t SZ = (size_t)BATCH * HEADS * NQ * DHEAD;   // 4M bf16 each
  bf16* Qb = (bf16*)d_ws;
  bf16* Kb = Qb + SZ;
  bf16* Vb = Kb + SZ;
  bf16* Ob = Vb + SZ;

  const dim3 blk(256);
  const dim3 gproj(INNER / BN, (BATCH * NQ) / BM);        // (8, 32)

  gemm_bf16_wmma<float, 0><<<gproj, blk, 0, stream>>>(x,   Wq, nullptr, Qb, nullptr,
                                                      BATCH * NQ,  QDIM, INNER);
  gemm_bf16_wmma<float, 0><<<gproj, blk, 0, stream>>>(ctx, Wk, nullptr, Kb, nullptr,
                                                      BATCH * NKV, QDIM, INNER);
  gemm_bf16_wmma<float, 0><<<gproj, blk, 0, stream>>>(ctx, Wv, nullptr, Vb, nullptr,
                                                      BATCH * NKV, QDIM, INNER);

  attn_wmma<<<dim3(BATCH * HEADS * (NQ / QTILE)), blk, 0, stream>>>(
      Qb, Kb, Vb, tgt, srcm, Ob);

  gemm_bf16_wmma<bf16, 1><<<gproj, blk, 0, stream>>>(Ob, Wo, bo, nullptr, out,
                                                     BATCH * NQ, INNER, QDIM);
}
